// ConvSE3_63376537420064
// MI455X (gfx1250) — matrix operand order repaired and checked by reading, not compiled, not run on hardware
//
#include <hip/hip_runtime.h>

typedef __attribute__((ext_vector_type(16))) _Float16 v16h;
typedef __attribute__((ext_vector_type(8)))  float    v8f;

static constexpr int NN = 10000;   // nodes
static constexpr int NE = 100000;  // edges

// ---------------------------------------------------------------------------
// Host-side parameter pointer bundle (36 pointers, passed by value).
struct RadialParams {
  const float* W1[4]; const float* b1[4]; const float* g1[4]; const float* be1[4];
  const float* W2[4]; const float* b2[4]; const float* g2[4]; const float* be2[4];
  const float* b3[4];
};

// ---------------------------------------------------------------------------
__global__ void zero_f32(float* __restrict__ p, int n) {
  int i = blockIdx.x * blockDim.x + threadIdx.x;
  if (i < n) p[i] = 0.f;
}

__global__ void degree_kernel(const int* __restrict__ dst, float* __restrict__ deg) {
  int e = blockIdx.x * blockDim.x + threadIdx.x;
  if (e < NE) atomicAdd(&deg[dst[e]], 1.f);
}

// Pre-swizzle W3 (fp32, row-major [32, ncols]) into fp16 WMMA B-fragment order:
// pack[ct][lane][idx]  with  K = idx + 16*(lane>>4),  N = ct*16 + (lane&15).
// ctile map: pair00 -> ct 0..15, pair01 -> 16..31, pair10 -> 32..47, pair11 -> 48..95.
__global__ void pack_w3_kernel(const float* __restrict__ w3a, const float* __restrict__ w3b,
                               const float* __restrict__ w3c, const float* __restrict__ w3d,
                               _Float16* __restrict__ pack) {
  int idx = blockIdx.x * blockDim.x + threadIdx.x;
  if (idx >= 96 * 512) return;
  int ct   = idx >> 9;
  int lane = (idx >> 4) & 31;
  int h    = idx & 15;
  const float* W3; int ctp, ncols;
  if (ct < 16)      { W3 = w3a; ctp = ct;      ncols = 256; }
  else if (ct < 32) { W3 = w3b; ctp = ct - 16; ncols = 256; }
  else if (ct < 48) { W3 = w3c; ctp = ct - 32; ncols = 256; }
  else              { W3 = w3d; ctp = ct - 48; ncols = 768; }
  int k   = h + 16 * (lane >> 4);
  int col = ctp * 16 + (lane & 15);
  pack[idx] = (_Float16)W3[k * ncols + col];
}

// Self-interaction + degree normalization factor.  Summed over incoming edges the
// self term is deg[n]*selfW.h[n]; divided by max(deg,1) it is flag(deg>0)*selfW.h[n].
__global__ void node_self_kernel(const float* __restrict__ h0, const float* __restrict__ h1,
                                 const float* __restrict__ s0, const float* __restrict__ s1,
                                 const float* __restrict__ deg, float* __restrict__ invden,
                                 float* __restrict__ out0, float* __restrict__ out1) {
  int n = blockIdx.x * blockDim.x + threadIdx.x;
  if (n >= NN) return;
  float dg  = deg[n];
  float inv = 1.f / fmaxf(dg, 1.f);
  invden[n] = inv;
  float flag = dg > 0.f ? 1.f : 0.f;
  for (int o = 0; o < 16; ++o) {
    float a = 0.f;
    for (int c = 0; c < 16; ++c) a += s0[o * 16 + c] * h0[n * 16 + c];
    out0[n * 16 + o] = flag * a;
  }
  for (int o = 0; o < 16; ++o) {
    float a0 = 0.f, a1 = 0.f, a2 = 0.f;
    for (int c = 0; c < 16; ++c) {
      float wv = s1[o * 16 + c];
      a0 += wv * h1[(n * 16 + c) * 3 + 0];
      a1 += wv * h1[(n * 16 + c) * 3 + 1];
      a2 += wv * h1[(n * 16 + c) * 3 + 2];
    }
    out1[n * 48 + o * 3 + 0] = flag * a0;
    out1[n * 48 + o * 3 + 1] = flag * a1;
    out1[n * 48 + o * 3 + 2] = flag * a2;
  }
}

// ---------------------------------------------------------------------------
// LayerNorm + ReLU over a 32-wide row split across lane pairs (lane, lane^16),
// writing fp16 results into the WMMA A-fragment layout:
//   element (M=e16, K=c) -> lane e16 + 16*(c>>4), slot (c&15).
__device__ __forceinline__ void ln_relu_to_afrag(int e16, int hf, const float* vals,
                                                 float s, float ss,
                                                 const float* g, const float* be,
                                                 _Float16* afrag) {
  float mu  = s * (1.f / 32.f);
  float var = ss * (1.f / 32.f) - mu * mu;
  float rs  = rsqrtf(var + 1e-5f);
#pragma unroll
  for (int q = 0; q < 16; ++q) {
    int c = hf * 16 + q;
    float v = (vals[q] - mu) * rs * g[c] + be[c];
    v = fmaxf(v, 0.f);
    int grp = c >> 4;
    int id  = c & 15;
    afrag[(e16 + 16 * grp) * 16 + id] = (_Float16)v;
  }
}

// One (t_in,t_out) tensor-product block for a 16-edge tile handled by one wave.
//   P=0:(0,0)  P=1:(0,1)  P=2:(1,0)  P=3:(1,1)
template <int P>
__device__ __forceinline__ void process_pair(
    int lane, int e16, int hf,
    const float* fvec, float bs00, const float* bas01,
    const float* sW1p, const float* sb1p, const float* sg1p, const float* sbe1p,
    const _Float16* sW2fp, const float* sb2p, const float* sg2p, const float* sbe2p,
    const float* sb3, const _Float16* __restrict__ w3pack,
    float* scr, _Float16* afrag,
    const float4* h0v, const float4* t10v, const float4* t11v,
    float* m0, float* m1) {
  constexpr int NT  = (P == 3) ? 3 : 1;   // ctiles per output channel o
  constexpr int CTB = P * 16;             // ctile base in w3pack
  constexpr int B3B = P * 256;            // b3 base offset

  // ---- layer 1 (5 -> 32), scalar ----
  float vals[16];
  float s = 0.f, ss = 0.f;
#pragma unroll
  for (int q = 0; q < 16; ++q) {
    int c = hf * 16 + q;
    float v = sb1p[c];
#pragma unroll
    for (int j = 0; j < 5; ++j) v += fvec[j] * sW1p[j * 32 + c];
    vals[q] = v; s += v; ss += v * v;
  }
  s  += __shfl_xor(s, 16);
  ss += __shfl_xor(ss, 16);
  ln_relu_to_afrag(e16, hf, vals, s, ss, sg1p, sbe1p, afrag);

  // ---- layer 2 (32 -> 32) via WMMA ----
  v16h aF = *(const v16h*)(afrag + lane * 16);
  int n  = lane & 15;
  int er = 8 * (lane >> 4);
#pragma unroll
  for (int ct = 0; ct < 2; ++ct) {
    v16h bF = *(const v16h*)(sW2fp + (ct * 32 + lane) * 16);
    v8f z = {0.f, 0.f, 0.f, 0.f, 0.f, 0.f, 0.f, 0.f};
    v8f d = __builtin_amdgcn_wmma_f32_16x16x32_f16(false, aF, false, bF, (short)0, z,
                                                   false, false);
    float b2v = sb2p[ct * 16 + n];
#pragma unroll
    for (int r = 0; r < 8; ++r) scr[(er + r) * 32 + ct * 16 + n] = d[r] + b2v;
  }
  float s2 = 0.f, ss2 = 0.f;
#pragma unroll
  for (int q = 0; q < 16; ++q) {
    int c = hf * 16 + q;
    float v = scr[e16 * 32 + c];
    vals[q] = v; s2 += v; ss2 += v * v;
  }
  s2  += __shfl_xor(s2, 16);
  ss2 += __shfl_xor(ss2, 16);
  ln_relu_to_afrag(e16, hf, vals, s2, ss2, sg2p, sbe2p, afrag);
  v16h aF2 = *(const v16h*)(afrag + lane * 16);

  const float4* scrv = (const float4*)scr;

  // ---- layer 3 (32 -> nf*256) via WMMA, consumed per output channel o ----
  for (int o = 0; o < 16; ++o) {
    // keep next B-fragment strip ahead of the tensor pipe
    if (o + 1 < 16)
      __builtin_prefetch(w3pack + ((size_t)(CTB + (o + 1) * NT) * 512 + lane * 16), 0, 3);
#pragma unroll
    for (int tt = 0; tt < NT; ++tt) {
      int ct = CTB + o * NT + tt;
      v16h bF = *(const v16h*)(w3pack + ((size_t)ct * 512 + lane * 16));
      v8f z = {0.f, 0.f, 0.f, 0.f, 0.f, 0.f, 0.f, 0.f};
      v8f d = __builtin_amdgcn_wmma_f32_16x16x32_f16(false, aF2, false, bF, (short)0, z,
                                                     false, false);
      float b3v = sb3[B3B + (o * NT + tt) * 16 + n];
#pragma unroll
      for (int r = 0; r < 8; ++r) scr[(er + r) * (NT * 16) + tt * 16 + n] = d[r] + b3v;
    }
    // ---- contract the staged R slab (16 edges x NT*16) with tmp tensors ----
    if constexpr (P != 3) {
      float4 sA = scrv[e16 * 4 + hf * 2 + 0];
      float4 sB = scrv[e16 * 4 + hf * 2 + 1];
      float4 tA, tB;
      if constexpr (P == 2) {
        tA = t10v[e16 * 4 + hf * 2 + 0];
        tB = t10v[e16 * 4 + hf * 2 + 1];
      } else {
        tA = h0v[e16 * 4 + hf * 2 + 0];
        tB = h0v[e16 * 4 + hf * 2 + 1];
      }
      float part = sA.x * tA.x + sA.y * tA.y + sA.z * tA.z + sA.w * tA.w +
                   sB.x * tB.x + sB.y * tB.y + sB.z * tB.z + sB.w * tB.w;
      part += __shfl_xor(part, 16);
      if constexpr (P == 0) {
        if (lane < 16) m0[e16 * 16 + o] += bs00 * part;
      } else if constexpr (P == 1) {
        if (lane < 16) {
          m1[e16 * 48 + o * 3 + 0] += part * bas01[0];
          m1[e16 * 48 + o * 3 + 1] += part * bas01[1];
          m1[e16 * 48 + o * 3 + 2] += part * bas01[2];
        }
      } else {
        if (lane < 16) m0[e16 * 16 + o] += part;
      }
    } else {
      // R slab row e16: 48 floats; this lane consumes i in [8*hf, 8*hf+8)
      float sval[24];
#pragma unroll
      for (int j = 0; j < 6; ++j) {
        float4 v = scrv[e16 * 12 + hf * 6 + j];
        sval[j * 4 + 0] = v.x; sval[j * 4 + 1] = v.y;
        sval[j * 4 + 2] = v.z; sval[j * 4 + 3] = v.w;
      }
      float p0 = 0.f, p1 = 0.f, p2 = 0.f;
#pragma unroll
      for (int q = 0; q < 8; ++q) {
        int i = hf * 8 + q;
        float4 t0 = t11v[(e16 * 16 + i) * 3 + 0];   // d0f0 d0f1 d0f2 d1f0
        float4 t1 = t11v[(e16 * 16 + i) * 3 + 1];   // d1f1 d1f2 d2f0 d2f1
        float4 t2 = t11v[(e16 * 16 + i) * 3 + 2];   // d2f2  -    -    -
        float r0 = sval[q * 3 + 0];
        float r1 = sval[q * 3 + 1];
        float r2 = sval[q * 3 + 2];
        p0 += r0 * t0.x + r1 * t0.y + r2 * t0.z;
        p1 += r0 * t0.w + r1 * t1.x + r2 * t1.y;
        p2 += r0 * t1.z + r1 * t1.w + r2 * t2.x;
      }
      p0 += __shfl_xor(p0, 16);
      p1 += __shfl_xor(p1, 16);
      p2 += __shfl_xor(p2, 16);
      if (lane < 16) {
        m1[e16 * 48 + o * 3 + 0] += p0;
        m1[e16 * 48 + o * 3 + 1] += p1;
        m1[e16 * 48 + o * 3 + 2] += p2;
      }
    }
  }
}

// ---------------------------------------------------------------------------
// 64 threads = 2 wave32's; each wave owns a 16-edge tile (M dim of the WMMAs).
__global__ __launch_bounds__(64) void edge_kernel(
    const float* __restrict__ h0, const float* __restrict__ h1,
    const float* __restrict__ wE, const float* __restrict__ rE,
    const float* __restrict__ b00, const float* __restrict__ b01,
    const float* __restrict__ b10, const float* __restrict__ b11,
    const int* __restrict__ src, const int* __restrict__ dst,
    RadialParams prm, const _Float16* __restrict__ w3pack,
    const float* __restrict__ invden,
    float* __restrict__ out0, float* __restrict__ out1) {
  __shared__ float sW1[4][160];
  __shared__ float sb1[4][32], sg1[4][32], sbe1[4][32];
  __shared__ __align__(32) _Float16 sW2f[4][1024];  // pre-swizzled B fragments
  __shared__ float sb2[4][32], sg2[4][32], sbe2[4][32];
  __shared__ float sb3[1536];
  __shared__ __align__(16) float scr[2][768];        // staging (x2 raw / R slab)
  __shared__ __align__(32) _Float16 afr[2][512];     // A fragments (16x32 fp16)
  __shared__ float4 h0s[2][64];                      // h0[src], 4 chunks/edge
  __shared__ float4 t10[2][64];                      // sum_k basis10*h1src
  __shared__ float4 t11[2][768];                     // [e][i][3 chunks] (9 used of 12)
  __shared__ float m0[2][256];                       // msg accum type-0
  __shared__ float m1[2][768];                       // msg accum type-1

  int tid = threadIdx.x;
  // ---- stage weights into LDS (whole block) ----
  for (int q = tid; q < 640; q += 64) ((float*)sW1)[q] = prm.W1[q / 160][q % 160];
  for (int q = tid; q < 128; q += 64) {
    int p_ = q >> 5, c = q & 31;
    sb1[p_][c] = prm.b1[p_][c];  sg1[p_][c] = prm.g1[p_][c];  sbe1[p_][c] = prm.be1[p_][c];
    sb2[p_][c] = prm.b2[p_][c];  sg2[p_][c] = prm.g2[p_][c];  sbe2[p_][c] = prm.be2[p_][c];
  }
  for (int q = tid; q < 4096; q += 64) {  // W2 -> fp16 B-fragment layout
    int p_ = q >> 10, rem = q & 1023;
    int ct = rem >> 9, r2 = rem & 511;
    int ln = r2 >> 4,  hh = r2 & 15;
    int k   = hh + 16 * (ln >> 4);
    int col = ct * 16 + (ln & 15);
    ((_Float16*)sW2f)[q] = (_Float16)prm.W2[p_][k * 32 + col];
  }
  for (int q = tid; q < 1536; q += 64) {
    int p_ = q >> 8; if (p_ > 3) p_ = 3;
    sb3[q] = prm.b3[p_][q - p_ * 256];
  }
  __syncthreads();

  int lane = tid & 31;
  int wv   = tid >> 5;
  int e16  = lane & 15;
  int hf   = lane >> 4;
  int eg   = blockIdx.x * 32 + wv * 16 + e16;
  bool valid = eg < NE;
  int ec = valid ? eg : NE - 1;
  int sE = src[ec], dE = dst[ec];

  float fvec[5];
#pragma unroll
  for (int j = 0; j < 4; ++j) fvec[j] = wE[ec * 4 + j];
  fvec[4] = rE[ec];
  float bs00 = b00[ec];
  float bas01[3];
#pragma unroll
  for (int d = 0; d < 3; ++d) bas01[d] = b01[ec * 3 + d];

  // ---- gather source features; precompute tmp[i,d,f] = sum_k basis*h_src ----
  float4* H0v  = &h0s[wv][0];
  float4* T10v = &t10[wv][0];
  float4* T11v = &t11[wv][0];
  {
    const float4* h0g = (const float4*)(h0 + (size_t)sE * 16);
    H0v[e16 * 4 + hf * 2 + 0] = h0g[hf * 2 + 0];
    H0v[e16 * 4 + hf * 2 + 1] = h0g[hf * 2 + 1];
    // 24 contiguous floats of h1[src] for this lane's 8 i-values
    const float4* h1g = (const float4*)(h1 + (size_t)sE * 48 + hf * 24);
    float hv[24];
#pragma unroll
    for (int j = 0; j < 6; ++j) {
      float4 t = h1g[j];
      hv[j * 4 + 0] = t.x; hv[j * 4 + 1] = t.y; hv[j * 4 + 2] = t.z; hv[j * 4 + 3] = t.w;
    }
    float bk0 = b10[ec * 3 + 0], bk1 = b10[ec * 3 + 1], bk2 = b10[ec * 3 + 2];
    float bv[27];
#pragma unroll
    for (int j = 0; j < 27; ++j) bv[j] = b11[ec * 27 + j];
    float t10loc[8];
#pragma unroll
    for (int q = 0; q < 8; ++q) {
      int i = hf * 8 + q;
      float hk0 = hv[q * 3 + 0], hk1 = hv[q * 3 + 1], hk2 = hv[q * 3 + 2];
      t10loc[q] = bk0 * hk0 + bk1 * hk1 + bk2 * hk2;
      float tv[9];
#pragma unroll
      for (int d = 0; d < 3; ++d)
#pragma unroll
        for (int f = 0; f < 3; ++f)
          tv[d * 3 + f] = bv[d * 9 + 0 * 3 + f] * hk0 +
                          bv[d * 9 + 1 * 3 + f] * hk1 +
                          bv[d * 9 + 2 * 3 + f] * hk2;
      T11v[(e16 * 16 + i) * 3 + 0] = make_float4(tv[0], tv[1], tv[2], tv[3]);
      T11v[(e16 * 16 + i) * 3 + 1] = make_float4(tv[4], tv[5], tv[6], tv[7]);
      T11v[(e16 * 16 + i) * 3 + 2] = make_float4(tv[8], 0.f, 0.f, 0.f);
    }
    T10v[e16 * 4 + hf * 2 + 0] = make_float4(t10loc[0], t10loc[1], t10loc[2], t10loc[3]);
    T10v[e16 * 4 + hf * 2 + 1] = make_float4(t10loc[4], t10loc[5], t10loc[6], t10loc[7]);
  }
  if (lane < 16) {
    for (int o = 0; o < 16; ++o) m0[wv][e16 * 16 + o] = 0.f;
    for (int j = 0; j < 48; ++j) m1[wv][e16 * 48 + j] = 0.f;
  }

  // ---- four (t_in,t_out) tensor-product blocks ----
  process_pair<0>(lane, e16, hf, fvec, bs00, bas01, sW1[0], sb1[0], sg1[0], sbe1[0],
                  sW2f[0], sb2[0], sg2[0], sbe2[0], sb3, w3pack, scr[wv], afr[wv],
                  H0v, T10v, T11v, &m0[wv][0], &m1[wv][0]);
  process_pair<1>(lane, e16, hf, fvec, bs00, bas01, sW1[1], sb1[1], sg1[1], sbe1[1],
                  sW2f[1], sb2[1], sg2[1], sbe2[1], sb3, w3pack, scr[wv], afr[wv],
                  H0v, T10v, T11v, &m0[wv][0], &m1[wv][0]);
  process_pair<2>(lane, e16, hf, fvec, bs00, bas01, sW1[2], sb1[2], sg1[2], sbe1[2],
                  sW2f[2], sb2[2], sg2[2], sbe2[2], sb3, w3pack, scr[wv], afr[wv],
                  H0v, T10v, T11v, &m0[wv][0], &m1[wv][0]);
  process_pair<3>(lane, e16, hf, fvec, bs00, bas01, sW1[3], sb1[3], sg1[3], sbe1[3],
                  sW2f[3], sb2[3], sg2[3], sbe2[3], sb3, w3pack, scr[wv], afr[wv],
                  H0v, T10v, T11v, &m0[wv][0], &m1[wv][0]);

  // ---- scatter (mean-normalized) ----
  if (valid) {
    float inv = invden[dE];
    if (lane < 16) {
      for (int o = 0; o < 16; ++o)
        atomicAdd(&out0[dE * 16 + o], m0[wv][e16 * 16 + o] * inv);
    } else {
      for (int j = 0; j < 48; ++j)
        atomicAdd(&out1[dE * 48 + j], m1[wv][e16 * 48 + j] * inv);
    }
  }
}

// ---------------------------------------------------------------------------
extern "C" void kernel_launch(void* const* d_in, const int* in_sizes, int n_in,
                              void* d_out, int out_size, void* d_ws, size_t ws_size,
                              hipStream_t stream) {
  (void)n_in; (void)out_size; (void)ws_size;
  // Auto-detect flattening order (insertion vs sorted dict keys) from sizes.
  bool topSorted = (in_sizes[0] == NE);  // sorted: basis_00 (E) first; else h0 (N*16)
  const float *h0, *h1, *wE, *rE, *b00, *b01, *b10, *b11, *self0, *self1;
  const int *src, *dst;
  int pbase;
  if (!topSorted) {
    h0  = (const float*)d_in[0];  h1  = (const float*)d_in[1];
    wE  = (const float*)d_in[2];  rE  = (const float*)d_in[3];
    b00 = (const float*)d_in[4];  b01 = (const float*)d_in[5];
    b10 = (const float*)d_in[6];  b11 = (const float*)d_in[7];
    src = (const int*)d_in[8];    dst = (const int*)d_in[9];
    pbase = 10;
    self0 = (const float*)d_in[50]; self1 = (const float*)d_in[51];
  } else {
    b00 = (const float*)d_in[0];  b01 = (const float*)d_in[1];
    b10 = (const float*)d_in[2];  b11 = (const float*)d_in[3];
    dst = (const int*)d_in[4];
    h0  = (const float*)d_in[5];  h1  = (const float*)d_in[6];
    pbase = 7;
    self0 = (const float*)d_in[47]; self1 = (const float*)d_in[48];
    rE  = (const float*)d_in[49];
    src = (const int*)d_in[50];
    wE  = (const float*)d_in[51];
  }
  bool innerSorted = (in_sizes[pbase + 1] == 1024);  // sorted puts W2 at +1
  int oW1, oB1, oG1, oBe1, oW2, oB2, oG2, oBe2, oW3, oB3;
  if (!innerSorted) { oW1=0; oB1=1; oG1=2; oBe1=3; oW2=4; oB2=5; oG2=6; oBe2=7; oW3=8; oB3=9; }
  else              { oW1=0; oW2=1; oW3=2; oB1=3; oB2=4; oB3=5; oBe1=6; oBe2=7; oG1=8; oG2=9; }

  RadialParams prm;
  const float* w3[4];
  for (int p = 0; p < 4; ++p) {
    int b = pbase + p * 10;
    prm.W1[p]  = (const float*)d_in[b + oW1];  prm.b1[p] = (const float*)d_in[b + oB1];
    prm.g1[p]  = (const float*)d_in[b + oG1];  prm.be1[p] = (const float*)d_in[b + oBe1];
    prm.W2[p]  = (const float*)d_in[b + oW2];  prm.b2[p] = (const float*)d_in[b + oB2];
    prm.g2[p]  = (const float*)d_in[b + oG2];  prm.be2[p] = (const float*)d_in[b + oBe2];
    prm.b3[p]  = (const float*)d_in[b + oB3];  w3[p] = (const float*)d_in[b + oW3];
  }

  // ws layout: deg [0,40000) | invden [40960,80960) | w3pack fp16 [81920,180224)
  float*     deg    = (float*)d_ws;
  float*     invden = (float*)((char*)d_ws + 40960);
  _Float16*  w3pack = (_Float16*)((char*)d_ws + 81920);
  float* out0 = (float*)d_out;
  float* out1 = out0 + NN * 16;

  zero_f32<<<(NN + 255) / 256, 256, 0, stream>>>(deg, NN);
  degree_kernel<<<(NE + 255) / 256, 256, 0, stream>>>(dst, deg);
  pack_w3_kernel<<<(96 * 512 + 255) / 256, 256, 0, stream>>>(w3[0], w3[1], w3[2], w3[3], w3pack);
  node_self_kernel<<<(NN + 127) / 128, 128, 0, stream>>>(h0, h1, self0, self1, deg,
                                                         invden, out0, out1);
  edge_kernel<<<(NE + 31) / 32, 64, 0, stream>>>(h0, h1, wE, rE, b00, b01, b10, b11,
                                                 src, dst, prm, w3pack, invden, out0, out1);
}
